// DeformableTransformerEncoderLayer_30485677867886
// MI455X (gfx1250) — matrix operand order repaired and checked by reading, not compiled, not run on hardware
//
#include <hip/hip_runtime.h>
#include <hip/hip_bf16.h>
#include <stdint.h>

// ---------------------------------------------------------------------------
// Deformable Transformer Encoder Layer for MI455X (gfx1250, wave32).
// GEMMs: bf16 WMMA (v_wmma_f32_16x16x32_bf16), fp32 accumulate,
//        double-buffered LDS fed by GLOBAL_LOAD_ASYNC_TO_LDS_B128 (ASYNCcnt).
// Sampling: wave-per-(b,l,head), lane = channel, L2-resident gathers.
// ---------------------------------------------------------------------------

typedef __attribute__((ext_vector_type(16))) __bf16 v16bf;
typedef __attribute__((ext_vector_type(8)))  float  v8f;

#define D_MODEL 256
#define NHEAD   8
#define HDIM    32
#define NLEV    4
#define NPT     4
#define LTOT    21760
#define BATCH   2

// GEMM tiling: 256 threads = 8 waves; block tile 128(M) x 64(N); K step 32.
// Each wave owns a 2x2 grid of 16x16 WMMA tiles (4 accumulators).
#define TM 128
#define TN 64
#define TK 32
#define AST 40                    // LDS row stride (halfs), padded
#define BST 40
#define A_HALFS (TM * AST)        // 5120 halfs = 10240 B per buffer
#define B_HALFS (TN * BST)        // 2560 halfs =  5120 B per buffer
#define BUF_HALFS (A_HALFS + B_HALFS)
#define BUF_BYTES (BUF_HALFS * 2)

__device__ __forceinline__ unsigned short f2bf(float f) {
    unsigned int u = __float_as_uint(f);
    if ((u & 0x7fffffffu) > 0x7f800000u) return (unsigned short)((u >> 16) | 0x0040u);
    return (unsigned short)((u + 0x7fffu + ((u >> 16) & 1u)) >> 16);
}

// f32 -> bf16 bits
__global__ void cvt_bf16_kernel(const float* __restrict__ in,
                                unsigned short* __restrict__ out, int n) {
    int i = blockIdx.x * 256 + threadIdx.x;
    if (i < n) out[i] = f2bf(in[i]);
}

// f32 [K][N] -> bf16 bits transposed [N][K] (weights; done once, tiny)
__global__ void cvt_bf16_t_kernel(const float* __restrict__ in,
                                  unsigned short* __restrict__ out, int K, int N) {
    int o = blockIdx.x * 256 + threadIdx.x;
    if (o < N * K) {
        int n = o / K, k = o - n * K;
        out[o] = f2bf(in[(size_t)k * N + n]);
    }
}

// src -> bf16 ; (src+pos) -> bf16
__global__ void prep_kernel(const float* __restrict__ src, const float* __restrict__ pos,
                            unsigned short* __restrict__ srcb, unsigned short* __restrict__ qb,
                            size_t n) {
    size_t i = (size_t)blockIdx.x * 256 + threadIdx.x;
    if (i < n) {
        float s = src[i];
        srcb[i] = f2bf(s);
        qb[i]   = f2bf(s + pos[i]);
    }
}

// C[M,N] = A[M,K] @ Bt[N,K]^T + bias ; optional relu, row-mask, f32/bf16 out.
// A row-major bf16 bits, Bt = B transposed (N-major, K contiguous) bf16 bits.
__global__ __launch_bounds__(256) void gemm_wmma_bf16(
    const unsigned short* __restrict__ A, const unsigned short* __restrict__ Bt,
    const float* __restrict__ bias,
    float* __restrict__ Cf, unsigned short* __restrict__ Cb,
    const unsigned char* __restrict__ rowmask,
    int M, int N, int K, int relu) {
    __shared__ __align__(16) unsigned short smem[2 * BUF_HALFS];

    const int n0 = blockIdx.x * TN;
    const int m0 = blockIdx.y * TM;
    const int t  = threadIdx.x;
    const int w  = t >> 5, l = t & 31;
    const int lm = l & 15, lh = l >> 4;
    // wave 2x2 tile origin: 4 wave-groups along M (32 rows each), 2 along N
    const int tm0 = (w & 3) * 32, tm1 = tm0 + 16;
    const int tn0 = (w >> 2) * 32, tn1 = tn0 + 16;

    const unsigned lds0 = (unsigned)(uintptr_t)(&smem[0]);  // LDS byte offset

    // Issue one K-slab (A: 512 16B chunks, B: 256 16B chunks) via async DMA.
    // 8 waves x 3 instructions x 32 lanes x 16B = 12 KB per slab.
    auto issue_stage = [&](int k0, int buf) {
        const unsigned abase = lds0 + (unsigned)buf * BUF_BYTES;
        const unsigned bbase = abase + A_HALFS * 2;
#pragma unroll
        for (int j = 0; j < 3; ++j) {
            const int g = w * 3 + j;            // chunk group 0..23 (uniform/wave)
            if (g < 16) {                       // A: 128 rows x 32 halfs
                const int c = g * 32 + l;       // 0..511
                const int row = c >> 2, ch = c & 3;
                const unsigned lds = abase + (unsigned)(row * (AST * 2) + ch * 16);
                const unsigned go  = (unsigned)(((size_t)(m0 + row) * K + k0) * 2) + ch * 16;
                asm volatile("global_load_async_to_lds_b128 %0, %1, %2"
                             :: "v"(lds), "v"(go), "s"(A) : "memory");
            } else {                            // B: 128 n-rows x 16 halfs (K-contig)
                const int c = (g - 16) * 32 + l;  // 0..255
                const int n = c >> 1, ch = c & 1;
                const unsigned lds = bbase + (unsigned)(n * (BST * 2) + ch * 16);
                const unsigned go  = (unsigned)(((size_t)(n0 + n) * K + k0) * 2) + ch * 16;
                asm volatile("global_load_async_to_lds_b128 %0, %1, %2"
                             :: "v"(lds), "v"(go), "s"(Bt) : "memory");
            }
        }
    };

    v8f c00 = {}, c01 = {}, c10 = {}, c11 = {};
    const int steps = K / TK;

    issue_stage(0, 0);
    for (int i = 0; i < steps; ++i) {
        const int buf = i & 1;
        if (i + 1 < steps) {
            issue_stage((i + 1) * TK, buf ^ 1);
            asm volatile("s_wait_asynccnt 0x3" ::: "memory");  // retire older slab
        } else {
            asm volatile("s_wait_asynccnt 0x0" ::: "memory");
        }
        __syncthreads();

        const unsigned short* Abuf = smem + buf * BUF_HALFS;
        const unsigned short* Bbuf = Abuf + A_HALFS;

        union { uint4 u[2]; v16bf v; } a0, a1, b0, b1;
        // A fragment: halfs 0-7 at K=lh*8, halfs 8-15 at K=16+lh*8 (ISA layout)
        a0.u[0] = *(const uint4*)(&Abuf[(tm0 + lm) * AST + lh * 8]);
        a0.u[1] = *(const uint4*)(&Abuf[(tm0 + lm) * AST + 16 + lh * 8]);
        a1.u[0] = *(const uint4*)(&Abuf[(tm1 + lm) * AST + lh * 8]);
        a1.u[1] = *(const uint4*)(&Abuf[(tm1 + lm) * AST + 16 + lh * 8]);
        // B fragment: lane = col, halfs 0-7 at K=lh*8, halfs 8-15 at K=16+lh*8
        b0.u[0] = *(const uint4*)(&Bbuf[(tn0 + lm) * BST + lh * 8]);
        b0.u[1] = *(const uint4*)(&Bbuf[(tn0 + lm) * BST + 16 + lh * 8]);
        b1.u[0] = *(const uint4*)(&Bbuf[(tn1 + lm) * BST + lh * 8]);
        b1.u[1] = *(const uint4*)(&Bbuf[(tn1 + lm) * BST + 16 + lh * 8]);

        c00 = __builtin_amdgcn_wmma_f32_16x16x32_bf16(false, a0.v, false, b0.v, (short)0, c00, false, false);
        c01 = __builtin_amdgcn_wmma_f32_16x16x32_bf16(false, a0.v, false, b1.v, (short)0, c01, false, false);
        c10 = __builtin_amdgcn_wmma_f32_16x16x32_bf16(false, a1.v, false, b0.v, (short)0, c10, false, false);
        c11 = __builtin_amdgcn_wmma_f32_16x16x32_bf16(false, a1.v, false, b1.v, (short)0, c11, false, false);
        __syncthreads();
    }

    // Epilogue: 4 tiles; lane holds (m = tile_m + lh*8 + r, n = tile_n + lm).
    auto store_tile = [&](v8f acc, int tm, int tn) {
        const int n = n0 + tn + lm;
        const float bvv = bias ? bias[n] : 0.0f;
#pragma unroll
        for (int r = 0; r < 8; ++r) {
            const int m = m0 + tm + lh * 8 + r;
            float v = acc[r] + bvv;
            if (relu) v = fmaxf(v, 0.0f);
            if (rowmask && rowmask[m]) v = 0.0f;
            const size_t o = (size_t)m * N + n;
            if (Cf) Cf[o] = v;
            if (Cb) Cb[o] = f2bf(v);
        }
    };
    store_tile(c00, tm0, tn0);
    store_tile(c01, tm0, tn1);
    store_tile(c10, tm1, tn0);
    store_tile(c11, tm1, tn1);
}

// softmax over 16 (level,point) logits per (b,l,head)
__global__ void softmax16_kernel(const float* __restrict__ in,
                                 float* __restrict__ out, int rows) {
    int r = blockIdx.x * 256 + threadIdx.x;
    if (r >= rows) return;
    const float* p = in + (size_t)r * 16;
    float m = -1e30f;
#pragma unroll
    for (int i = 0; i < 16; ++i) m = fmaxf(m, p[i]);
    float e[16], s = 0.0f;
#pragma unroll
    for (int i = 0; i < 16; ++i) { e[i] = __expf(p[i] - m); s += e[i]; }
    const float inv = 1.0f / s;
    float* q = out + (size_t)r * 16;
#pragma unroll
    for (int i = 0; i < 16; ++i) q[i] = e[i] * inv;
}

// Deformable sampling: one wave per (b, l, head); lane = channel (HD=32).
__global__ __launch_bounds__(256) void sample_kernel(
    const float* __restrict__ value, const float* __restrict__ off,
    const float* __restrict__ attnw, const float* __restrict__ refp,
    unsigned short* __restrict__ outb) {
    const int wid  = (blockIdx.x * 256 + threadIdx.x) >> 5;
    const int lane = threadIdx.x & 31;
    const int h = wid & 7;
    const int l = (wid >> 3) % LTOT;
    const int b = wid / (8 * LTOT);
    if (b >= BATCH) return;

    const int HL[4] = {128, 64, 32, 16};
    const int WL[4] = {128, 64, 32, 16};
    const int ST[4] = {0, 16384, 20480, 21504};

    const size_t row  = (size_t)b * LTOT + l;
    const float* offr = off   + row * D_MODEL + h * (NLEV * NPT * 2);
    const float* awr  = attnw + row * (NHEAD * NLEV * NPT) + h * (NLEV * NPT);
    const float* rp   = refp  + row * (NLEV * 2);
    const size_t vbase = (size_t)b * LTOT * D_MODEL + (size_t)h * HDIM + lane;

    float acc = 0.0f;
#pragma unroll
    for (int lv = 0; lv < NLEV; ++lv) {
        const int Wi = WL[lv], Hi = HL[lv], base = ST[lv];
        const float Wf = (float)Wi, Hf = (float)Hi;
        const float rx = rp[lv * 2 + 0], ry = rp[lv * 2 + 1];
#pragma unroll
        for (int p = 0; p < NPT; ++p) {
            const float ox = offr[(lv * NPT + p) * 2 + 0];
            const float oy = offr[(lv * NPT + p) * 2 + 1];
            const float aw = awr[lv * NPT + p];
            const float px = rx * Wf + ox - 0.5f;  // (rx + ox/W)*W - 0.5
            const float py = ry * Hf + oy - 0.5f;
            const float x0 = floorf(px), y0 = floorf(py);
            const float fx = px - x0, fy = py - y0;
            const int xi = (int)x0, yi = (int)y0;
            const float w00 = (1.f - fx) * (1.f - fy), w10 = fx * (1.f - fy);
            const float w01 = (1.f - fx) * fy,         w11 = fx * fy;
            auto corner = [&](int xx, int yy, float cw) {
                if (xx < 0 || xx >= Wi || yy < 0 || yy >= Hi) return;
                const int idx = base + yy * Wi + xx;
                acc += aw * cw * value[vbase + (size_t)idx * D_MODEL];
            };
            corner(xi,     yi,     w00);
            corner(xi + 1, yi,     w10);
            corner(xi,     yi + 1, w01);
            corner(xi + 1, yi + 1, w11);
        }
    }
    outb[row * D_MODEL + h * HDIM + lane] = f2bf(acc);
}

// y = LayerNorm(a + b) * g + beta ; fp32 out + optional bf16 out.
__global__ __launch_bounds__(256) void add_ln_kernel(
    const float* __restrict__ a, const float* __restrict__ b2,
    const float* __restrict__ g, const float* __restrict__ beta,
    float* __restrict__ outf, unsigned short* __restrict__ outb) {
    const int row = blockIdx.x;
    const int t = threadIdx.x;
    const size_t o = (size_t)row * D_MODEL + t;
    const float v = a[o] + b2[o];

    __shared__ float sm[256];
    __shared__ float smean, svar;
    sm[t] = v;
    __syncthreads();
    for (int s = 128; s > 0; s >>= 1) {
        if (t < s) sm[t] += sm[t + s];
        __syncthreads();
    }
    if (t == 0) smean = sm[0] * (1.0f / D_MODEL);
    __syncthreads();
    const float d = v - smean;
    sm[t] = d * d;
    __syncthreads();
    for (int s = 128; s > 0; s >>= 1) {
        if (t < s) sm[t] += sm[t + s];
        __syncthreads();
    }
    if (t == 0) svar = sm[0] * (1.0f / D_MODEL);
    __syncthreads();
    const float y = d * rsqrtf(svar + 1e-5f) * g[t] + beta[t];
    if (outf) outf[o] = y;
    if (outb) outb[o] = f2bf(y);
}

extern "C" void kernel_launch(void* const* d_in, const int* in_sizes, int n_in,
                              void* d_out, int out_size, void* d_ws, size_t ws_size,
                              hipStream_t stream) {
    (void)in_sizes; (void)n_in; (void)out_size; (void)ws_size;
    const float* src   = (const float*)d_in[0];
    const float* pos   = (const float*)d_in[1];
    const float* refp  = (const float*)d_in[2];
    const unsigned char* pmask = (const unsigned char*)d_in[5];
    const float* Wv    = (const float*)d_in[8];  const float* bv    = (const float*)d_in[9];
    const float* Woff  = (const float*)d_in[10]; const float* boff  = (const float*)d_in[11];
    const float* Wattn = (const float*)d_in[12]; const float* battn = (const float*)d_in[13];
    const float* Wout  = (const float*)d_in[14]; const float* bout  = (const float*)d_in[15];
    const float* g1    = (const float*)d_in[16]; const float* b1    = (const float*)d_in[17];
    const float* W1    = (const float*)d_in[18]; const float* bf1   = (const float*)d_in[19];
    const float* W2    = (const float*)d_in[20]; const float* bf2   = (const float*)d_in[21];
    const float* g3    = (const float*)d_in[22]; const float* b3    = (const float*)d_in[23];

    const size_t ML = (size_t)BATCH * LTOT;  // 43520 rows

    // ---- workspace carving --------------------------------------------------
    char* p = (char*)d_ws;
    auto carve = [&](size_t bytes) -> void* {
        void* r = (void*)p;
        p += (bytes + 255) & ~(size_t)255;
        return r;
    };
    unsigned short* Wv_t    = (unsigned short*)carve(256 * 256 * 2);   // [N][K]
    unsigned short* Woff_t  = (unsigned short*)carve(256 * 256 * 2);
    unsigned short* Wattn_t = (unsigned short*)carve(128 * 256 * 2);
    unsigned short* Wout_t  = (unsigned short*)carve(256 * 256 * 2);
    unsigned short* W1_t    = (unsigned short*)carve(1024 * 256 * 2);
    unsigned short* W2_t    = (unsigned short*)carve(256 * 1024 * 2);
    unsigned short* src_b   = (unsigned short*)carve(ML * 256 * 2);
    unsigned short* q_b     = (unsigned short*)carve(ML * 256 * 2);
    float*          valf    = (float*)carve(ML * 256 * 4);
    float*          offf    = (float*)carve(ML * 256 * 4);
    float*          logit   = (float*)carve(ML * 128 * 4);
    float*          attnw   = (float*)carve(ML * 128 * 4);
    unsigned short* samp_b  = (unsigned short*)carve(ML * 256 * 2);
    float*          src2    = (float*)carve(ML * 256 * 4);
    float*          xf      = (float*)carve(ML * 256 * 4);
    unsigned short* x_b     = (unsigned short*)carve(ML * 256 * 2);
    unsigned short* h_b     = (unsigned short*)carve(ML * 1024 * 2);
    float*          ff      = (float*)carve(ML * 256 * 4);

    // ---- weight conversions (transposed to [N][K] for async B staging) -----
    auto cvtT = [&](const float* in, unsigned short* out, int K, int N) {
        cvt_bf16_t_kernel<<<(N * K + 255) / 256, 256, 0, stream>>>(in, out, K, N);
    };
    cvtT(Wv, Wv_t, 256, 256);
    cvtT(Woff, Woff_t, 256, 256);
    cvtT(Wattn, Wattn_t, 256, 128);
    cvtT(Wout, Wout_t, 256, 256);
    cvtT(W1, W1_t, 256, 1024);
    cvtT(W2, W2_t, 1024, 256);

    // q = src + pos (bf16); src -> bf16
    prep_kernel<<<(unsigned)((ML * 256 + 255) / 256), 256, 0, stream>>>(
        src, pos, src_b, q_b, ML * 256);

    const unsigned GY = (unsigned)(ML / TM);  // 340
    auto gemm = [&](const unsigned short* A, const unsigned short* Bt,
                    const float* bias, float* Cf, unsigned short* Cb,
                    const unsigned char* mask, int N, int K, int relu) {
        dim3 g(N / TN, GY);
        gemm_wmma_bf16<<<g, 256, 0, stream>>>(A, Bt, bias, Cf, Cb, mask,
                                              (int)ML, N, K, relu);
    };

    // value = mask0(src @ Wv + bv)
    gemm(src_b, Wv_t, bv, valf, nullptr, pmask, 256, 256, 0);
    // offsets = q @ Woff + boff
    gemm(q_b, Woff_t, boff, offf, nullptr, nullptr, 256, 256, 0);
    // attention logits = q @ Wattn + battn
    gemm(q_b, Wattn_t, battn, logit, nullptr, nullptr, 128, 256, 0);

    // softmax over 16 per (b,l,head)
    {
        int rows = (int)(ML * NHEAD);
        softmax16_kernel<<<(rows + 255) / 256, 256, 0, stream>>>(logit, attnw, rows);
    }

    // deformable bilinear sampling -> bf16 [B,L,D]
    {
        size_t waves = ML * NHEAD;  // 348160
        sample_kernel<<<(unsigned)(waves * 32 / 256), 256, 0, stream>>>(
            valf, offf, attnw, refp, samp_b);
    }

    // src2 = samp @ Wout + bout
    gemm(samp_b, Wout_t, bout, src2, nullptr, nullptr, 256, 256, 0);
    // x = LN(src + src2)
    add_ln_kernel<<<(unsigned)ML, 256, 0, stream>>>(src, src2, g1, b1, xf, x_b);
    // h = relu(x @ W1 + bf1)   (bf16 only)
    gemm(x_b, W1_t, bf1, nullptr, h_b, nullptr, 1024, 256, 1);
    // ff = h @ W2 + bf2
    gemm(h_b, W2_t, bf2, ff, nullptr, nullptr, 256, 1024, 0);
    // out = LN(x + ff)
    add_ln_kernel<<<(unsigned)ML, 256, 0, stream>>>(xf, ff, g3, b3, (float*)d_out, nullptr);
}
